// MeSH_GCN_18253611008879
// MI455X (gfx1250) — compile-verified, matrix operand order checked by reading, not run on hardware
//
#include <hip/hip_runtime.h>
#include <hip/hip_bf16.h>
#include <math.h>

typedef __attribute__((ext_vector_type(16))) _Float16 v16h;
typedef __attribute__((ext_vector_type(8)))  float    v8f;
typedef __attribute__((ext_vector_type(2)))  _Float16 half2v;
typedef __attribute__((ext_vector_type(8)))  _Float16 half8v;
typedef __attribute__((ext_vector_type(4)))  unsigned u32x4;
typedef __attribute__((ext_vector_type(4)))  int      i32x4;
typedef __attribute__((ext_vector_type(8)))  int      i32x8;

typedef __attribute__((address_space(3))) void las_void;

#if defined(__has_builtin)
#  if __has_builtin(__builtin_amdgcn_tensor_load_to_lds)
#    define HAVE_TDM 1
#  endif
#endif
#ifndef HAVE_TDM
#  define HAVE_TDM 0
#endif

// ---- WMMA 16-bit fragment loader (CDNA5 ISA 7.12.2 A/B 16-bit layout) ----
// Element pairs (2v,2v+1) are consecutive K -> 8x 32-bit loads per fragment.
// Works on global or LDS (generic) pointers; requires 4-byte alignment.
__device__ __forceinline__ v16h load_frag_row(const _Float16* __restrict__ p,
                                              int hi8) {
  v16h f;
#pragma unroll
  for (int v = 0; v < 8; ++v) {
    int kk = ((v < 4) ? 2 * v : 16 + 2 * (v - 4)) + hi8;
    half2v t = *(const half2v*)(p + kk);
    f[2 * v]     = t.x;
    f[2 * v + 1] = t.y;
  }
  return f;
}

// ---------------------------------------------------------------- utilities
__global__ void k_zero_f32(float* p, long long n) {
  long long i = blockIdx.x * (long long)blockDim.x + threadIdx.x;
  if (i < n) p[i] = 0.0f;
}

__global__ void k_cvt_f16(const float* __restrict__ in, _Float16* __restrict__ out,
                          long long n) {
  long long i = blockIdx.x * (long long)blockDim.x + threadIdx.x;
  if (i < n) out[i] = (_Float16)in[i];
}

// ------------------------------------------------- embedding gather -> f16
// xf layout: [B=128][L=1024][C=208], channels 200..207 zero. 8 halfs/thread.
__global__ void k_embed_gather(const int* __restrict__ seq,
                               const float* __restrict__ emb,
                               _Float16* __restrict__ xf) {
  long long i = blockIdx.x * (long long)blockDim.x + threadIdx.x;
  const long long total = 128LL * 1024 * 26;
  if (i >= total) return;
  int c = (int)(i % 26) * 8;
  long long bl = i / 26;  // b*1024 + l
  half8v v = {};
  if (c < 200) {
    const float* ep = emb + (long long)seq[bl] * 200 + c;
#pragma unroll
    for (int j = 0; j < 8; ++j) v[j] = (_Float16)ep[j];
  }
  *(half8v*)(xf + bl * 208 + c) = v;
}

// --------------------------- pack conv weights [O,C,k] -> f16 [208][Kp]
// K ordered (j,c): kk = j*208 + c. With this order the im2col operand for
// output t is just x[b] + t*208 read contiguously over K (no index math).
__global__ void k_pack_convw(const float* __restrict__ w, _Float16* __restrict__ wp,
                             int ksz, int Kp) {
  long long i = blockIdx.x * (long long)blockDim.x + threadIdx.x;
  long long total = 208LL * Kp;
  if (i >= total) return;
  int kk = (int)(i % Kp);
  int o  = (int)(i / Kp);
  int j  = kk / 208;
  int c  = kk - j * 208;
  float v = 0.0f;
  if (o < 200 && c < 200 && j < ksz) v = w[((long long)o * 200 + c) * ksz + j];
  wp[i] = (_Float16)v;
}

__global__ void k_pack_W1(const float* __restrict__ W1, _Float16* __restrict__ w) {
  int i = blockIdx.x * blockDim.x + threadIdx.x;  // [224][512]
  if (i >= 224 * 512) return;
  int n = i % 512, kk = i / 512;
  w[i] = (_Float16)(kk < 200 ? W1[kk * 512 + n] : 0.0f);
}

__global__ void k_pack_W2(const float* __restrict__ W2, _Float16* __restrict__ w) {
  int i = blockIdx.x * blockDim.x + threadIdx.x;  // [512][208]
  if (i >= 512 * 208) return;
  int n = i % 208, kk = i / 208;
  w[i] = (_Float16)(n < 200 ? W2[kk * 200 + n] : 0.0f);
}

// --------------------------- conv-as-GEMM (NT form) via WMMA, fused epilogue
// y[o,t] = sum_kk wp[o][kk] * x[b][t*208 + kk]; pad-K weights are zero so the
// overlap tail contributes nothing. Epilogue: relu(y+bias), max over t via
// atomicMax on the uint pattern (post-relu values are >= 0).
__global__ void k_conv_wmma(const _Float16* __restrict__ xf,
                            const _Float16* __restrict__ wp,
                            const float* __restrict__ bias,
                            float* __restrict__ pool,
                            int Kp, int poolOff, int Tvalid) {
  int lane = threadIdx.x & 31, wave = threadIdx.x >> 5;
  int t0 = (blockIdx.x * 4 + wave) * 16;   // 64 t-tiles
  int o0 = blockIdx.y * 16;                // 13 o-tiles
  int b  = blockIdx.z;                     // 128 batches
  int hi8 = (lane & 16) ? 8 : 0;
  const _Float16* ap = wp + (long long)(o0 + (lane & 15)) * Kp;
  const _Float16* bp = xf + (long long)b * 1024 * 208 +
                       (long long)(t0 + (lane & 15)) * 208;
  v8f acc = {};
  for (int k0 = 0; k0 < Kp; k0 += 32) {
    __builtin_prefetch(ap + k0 + 128, 0, 3);
    __builtin_prefetch(bp + k0 + 128, 0, 3);
    v16h a  = load_frag_row(ap + k0, hi8);
    v16h bb = load_frag_row(bp + k0, hi8);
    acc = __builtin_amdgcn_wmma_f32_16x16x32_f16(false, a, false, bb,
                                                 (short)0, acc, false, false);
  }
  int t = t0 + (lane & 15);
#pragma unroll
  for (int r = 0; r < 8; ++r) {
    int o = o0 + r + hi8;
    if (o < 200 && t < Tvalid) {
      float v = acc[r] + bias[o];
      v = v > 0.0f ? v : 0.0f;
      atomicMax((unsigned int*)&pool[(long long)b * 416 + poolOff + o],
                __float_as_uint(v));
    }
  }
}

// ------------------------------------------------- GCN scatter-add kernels
__global__ void k_scatter1(const int* __restrict__ src, const int* __restrict__ dst,
                           const float* __restrict__ feat, float* __restrict__ agg1) {
  int e = blockIdx.x;
  int c = threadIdx.x;
  if (c >= 200) return;
  int s = src[e], d = dst[e];
  atomicAdd(&agg1[(long long)d * 224 + c], feat[(long long)s * 200 + c]);
}

__global__ void k_scatter2(const int* __restrict__ src, const int* __restrict__ dst,
                           const _Float16* __restrict__ h1, float* __restrict__ agg2) {
  int e = blockIdx.x;
  int s = src[e], d = dst[e];
  for (int c = threadIdx.x; c < 512; c += blockDim.x)
    atomicAdd(&agg2[(long long)d * 512 + c], (float)h1[(long long)s * 512 + c]);
}

// ---------------- LDS-staged WMMA GEMM, C = A[MxK] * B[KxN], f16 out
// Block: 128 threads = 4 waves sharing one 16-row A tile, spanning 64 cols.
// A tile staged by the Tensor Data Mover (one tensor_load_to_lds per K-step,
// D# LDS padding: 16-dword interval + 4-dword pad -> 40-half rows, bank-
// conflict-free reads). B tile staged coalesced + transposed into 34-half
// pitched rows (bank stride 17, coprime with 64).
__global__ void k_gemm_nn(const _Float16* __restrict__ A, int lda,
                          const _Float16* __restrict__ B, int ldb,
                          const float* __restrict__ bias,
                          _Float16* __restrict__ Ch, int ldc,
                          int M, int N, int Nvalid, int K, int relu) {
  __shared__ __align__(16) _Float16 As[16 * 40];
  __shared__ __align__(16) _Float16 Bst[64 * 34];
  int lane = threadIdx.x & 31, wave = threadIdx.x >> 5;
  int m0 = blockIdx.y * 16;
  int n0 = blockIdx.x * 64;
  int hi8 = (lane & 16) ? 8 : 0;

  v8f acc = {};
  for (int k0 = 0; k0 < K; k0 += 32) {
    if (k0) __syncthreads();   // previous compute done before LDS overwrite
    // ---- stage A tile [16 x 32] -> As (rows pitched to 40 halfs)
#if HAVE_TDM
    if (wave == 0) {
      unsigned lds = (unsigned)(size_t)(las_void*)&As[0];
      unsigned long long ga =
          (unsigned long long)(size_t)(A + (long long)m0 * lda + k0);
      u32x4 g0;
      g0[0] = 1u;                                   // count=1, user mode
      g0[1] = lds;                                  // lds_addr
      g0[2] = (unsigned)ga;                         // global_addr[31:0]
      g0[3] = ((unsigned)(ga >> 32) & 0x01FFFFFFu)  // global_addr[56:32]
              | 0x80000000u;                        // type=2 (image)
      i32x8 g1;
      g1[0] = (int)((1u << 16)      // data_size = 2 bytes
                    | (1u << 20)    // pad_enable
                    | (3u << 22)    // pad_interval: every 16 dwords (one row)
                    | (3u << 25));  // pad_amount: 4 dwords -> 40-half pitch
      g1[1] = (int)(32u << 16);     // tensor_dim0 = 32
      g1[2] = (int)(16u << 16);     // tensor_dim1 = 16
      g1[3] = (int)(32u << 16);     // tile_dim0 = 32
      g1[4] = 16;                   // tile_dim1 = 16 (tile_dim2 = 0)
      g1[5] = (int)(unsigned)lda;   // tensor_dim0_stride
      g1[6] = 0;
      g1[7] = 0;
      i32x4 z4 = {};
#if __clang_major__ >= 23
      i32x8 z8 = {};
      __builtin_amdgcn_tensor_load_to_lds(g0, g1, z4, z4, z8, 0);
#else
      __builtin_amdgcn_tensor_load_to_lds(g0, g1, z4, z4, 0);
#endif
      __builtin_amdgcn_s_wait_tensorcnt(0);
    }
#else
    if (threadIdx.x < 64) {
      int row = threadIdx.x >> 2, part = threadIdx.x & 3;
      *(uint4*)&As[row * 40 + part * 8] =
          *(const uint4*)(A + (long long)(m0 + row) * lda + k0 + part * 8);
    }
#endif
    // ---- stage B tile [32 x 64] coalesced, transposed -> Bst[n][kk]
    for (int q = threadIdx.x; q < 256; q += 128) {
      int kk = q >> 3, part = q & 7;
      int ng = n0 + part * 8;
      half8v val = {};
      if (ng < N) val = *(const half8v*)(B + (long long)(k0 + kk) * ldb + ng);
#pragma unroll
      for (int i2 = 0; i2 < 8; ++i2) Bst[(part * 8 + i2) * 34 + kk] = val[i2];
    }
    __syncthreads();
    v16h a  = load_frag_row(&As[(lane & 15) * 40], hi8);
    v16h bf = load_frag_row(&Bst[(wave * 16 + (lane & 15)) * 34], hi8);
    acc = __builtin_amdgcn_wmma_f32_16x16x32_f16(false, a, false, bf,
                                                 (short)0, acc, false, false);
  }
  int n = n0 + wave * 16 + (lane & 15);
  if (n < Nvalid) {
    float bv = bias ? bias[n] : 0.0f;
#pragma unroll
    for (int r = 0; r < 8; ++r) {
      int m = m0 + r + hi8;
      float v = acc[r] + bv;
      if (relu) v = v > 0.0f ? v : 0.0f;
      Ch[(long long)m * ldc + n] = (_Float16)v;
    }
  }
}

// ---------------- final bilinear: C = A[MxK] * B[NxK]^T, fused sigmoid
__global__ void k_gemm_nt_sig(const _Float16* __restrict__ A, int lda,
                              const _Float16* __restrict__ B, int ldb,
                              float* __restrict__ out, int ldo,
                              int M, int N, int K) {
  int lane = threadIdx.x & 31, wave = threadIdx.x >> 5;
  int nTile = blockIdx.x * 4 + wave;
  if (nTile * 16 >= N) return;     // wave-uniform, no barriers in kernel
  int m0 = blockIdx.y * 16, n0 = nTile * 16;
  int hi8 = (lane & 16) ? 8 : 0;
  const _Float16* ap = A + (long long)(m0 + (lane & 15)) * lda;
  const _Float16* bp = B + (long long)(n0 + (lane & 15)) * ldb;
  v8f acc = {};
  for (int k0 = 0; k0 < K; k0 += 32) {
    __builtin_prefetch(bp + k0 + 128, 0, 3);
    v16h a = load_frag_row(ap + k0, hi8);
    v16h b = load_frag_row(bp + k0, hi8);
    acc = __builtin_amdgcn_wmma_f32_16x16x32_f16(false, a, false, b,
                                                 (short)0, acc, false, false);
  }
  int n = n0 + (lane & 15);
#pragma unroll
  for (int r = 0; r < 8; ++r) {
    int m = m0 + r + hi8;
    out[(long long)m * ldo + n] = 1.0f / (1.0f + __expf(-acc[r]));
  }
}

// ------------------------------------------------ feature assembly helpers
__global__ void k_fill_lf(const float* __restrict__ feat, _Float16* __restrict__ lf) {
  long long i = blockIdx.x * (long long)blockDim.x + threadIdx.x;
  const long long total = 30000LL * 216;
  if (i >= total) return;
  int j = (int)(i % 216);
  long long nrow = i / 216;
  _Float16 v = (_Float16)0.0f;
  if (j < 200) v = (_Float16)feat[nrow * 200 + j];
  lf[nrow * 416 + 200 + j] = v;
}

__global__ void k_pool_to_f16(const float* __restrict__ pool,
                              _Float16* __restrict__ xf16) {
  int i = blockIdx.x * blockDim.x + threadIdx.x;
  if (i >= 128 * 416) return;
  int c = i % 416;
  xf16[i] = (_Float16)(c < 400 ? pool[i] : 0.0f);
}

// ---------------------------------------------------------------------------
extern "C" void kernel_launch(void* const* d_in, const int* in_sizes, int n_in,
                              void* d_out, int out_size, void* d_ws, size_t ws_size,
                              hipStream_t stream) {
  const int*   seq      = (const int*)d_in[0];
  const int*   src      = (const int*)d_in[1];
  const int*   dst      = (const int*)d_in[2];
  const float* features = (const float*)d_in[3];
  const float* emb      = (const float*)d_in[4];
  const float* w3       = (const float*)d_in[5];
  const float* b3       = (const float*)d_in[6];
  const float* w5       = (const float*)d_in[7];
  const float* b5       = (const float*)d_in[8];
  const float* W1       = (const float*)d_in[9];
  const float* b1       = (const float*)d_in[10];
  const float* W2       = (const float*)d_in[11];
  const float* b2       = (const float*)d_in[12];
  float* out = (float*)d_out;

  char* ws = (char*)d_ws;
  size_t off = 0;
  auto alloc = [&](size_t bytes) {
    char* p = ws + off;
    off = (off + bytes + 255) & ~(size_t)255;
    return p;
  };

  const long long XFN = 128LL * 1024 * 208;
  _Float16* xf    = (_Float16*)alloc((XFN + 1024) * 2);  // + overlap slack
  _Float16* w3p   = (_Float16*)alloc(208LL * 640 * 2);
  _Float16* w5p   = (_Float16*)alloc(208LL * 1056 * 2);
  float*    pool  = (float*)   alloc(128LL * 416 * 4);   // [p3|p5|pad]
  _Float16* xf16  = (_Float16*)alloc(128LL * 416 * 2);
  float*    agg1f = (float*)   alloc(30000LL * 224 * 4);
  _Float16* agg1h = (_Float16*)alloc(30000LL * 224 * 2);
  _Float16* W1h   = (_Float16*)alloc(224LL * 512 * 2);
  _Float16* h1h   = (_Float16*)alloc(30000LL * 512 * 2);
  float*    agg2f = (float*)   alloc(30000LL * 512 * 4);
  _Float16* agg2h = (_Float16*)alloc(30000LL * 512 * 2);
  _Float16* W2h   = (_Float16*)alloc(512LL * 208 * 2);
  _Float16* lf    = (_Float16*)alloc(30000LL * 416 * 2); // [h2|feat|pad]

  // zero accumulator scratch + x slack (every call; graph-replay safe)
  {
    long long n = 128LL * 416;
    k_zero_f32<<<dim3((unsigned)((n + 255) / 256)), 256, 0, stream>>>(pool, n);
    n = 30000LL * 224;
    k_zero_f32<<<dim3((unsigned)((n + 255) / 256)), 256, 0, stream>>>(agg1f, n);
    n = 30000LL * 512;
    k_zero_f32<<<dim3((unsigned)((n + 255) / 256)), 256, 0, stream>>>(agg2f, n);
    k_zero_f32<<<dim3(2), 256, 0, stream>>>((float*)(xf + XFN), 512);
  }

  // embedding gather + weight packing
  {
    long long n = 128LL * 1024 * 26;
    k_embed_gather<<<dim3((unsigned)((n + 255) / 256)), 256, 0, stream>>>(seq, emb, xf);
    n = 208LL * 640;
    k_pack_convw<<<dim3((unsigned)((n + 255) / 256)), 256, 0, stream>>>(w3, w3p, 3, 640);
    n = 208LL * 1056;
    k_pack_convw<<<dim3((unsigned)((n + 255) / 256)), 256, 0, stream>>>(w5, w5p, 5, 1056);
    k_pack_W1<<<dim3((224 * 512 + 255) / 256), 256, 0, stream>>>(W1, W1h);
    k_pack_W2<<<dim3((512 * 208 + 255) / 256), 256, 0, stream>>>(W2, W2h);
  }

  // CNN branch: conv(k=3), conv(k=5) as WMMA NT-GEMMs w/ fused relu+maxpool
  k_conv_wmma<<<dim3(16, 13, 128), 128, 0, stream>>>(xf, w3p, b3, pool, 640, 0, 1022);
  k_conv_wmma<<<dim3(16, 13, 128), 128, 0, stream>>>(xf, w5p, b5, pool, 1056, 200, 1020);
  k_pool_to_f16<<<dim3((128 * 416 + 255) / 256), 256, 0, stream>>>(pool, xf16);

  // GCN layer 1: scatter-add, cast, GEMM(+bias,relu) -> h1 (f16)
  k_scatter1<<<dim3(300000), 256, 0, stream>>>(src, dst, features, agg1f);
  {
    long long n = 30000LL * 224;
    k_cvt_f16<<<dim3((unsigned)((n + 255) / 256)), 256, 0, stream>>>(agg1f, agg1h, n);
  }
  k_gemm_nn<<<dim3(512 / 64, 30000 / 16), 128, 0, stream>>>(
      agg1h, 224, W1h, 512, b1, h1h, 512, 30000, 512, 512, 224, 1);

  // GCN layer 2: scatter-add on h1, cast, GEMM(+bias) -> lf cols [0,200)
  k_scatter2<<<dim3(300000), 256, 0, stream>>>(src, dst, h1h, agg2f);
  {
    long long n = 30000LL * 512;
    k_cvt_f16<<<dim3((unsigned)((n + 255) / 256)), 256, 0, stream>>>(agg2f, agg2h, n);
  }
  k_gemm_nn<<<dim3(4, 30000 / 16), 128, 0, stream>>>(
      agg2h, 512, W2h, 208, b2, lf, 416, 30000, 208, 200, 512, 0);

  // label_feature cols [200,416): raw features + zero pad
  {
    long long n = 30000LL * 216;
    k_fill_lf<<<dim3((unsigned)((n + 255) / 256)), 256, 0, stream>>>(features, lf);
  }

  // bilinear match + sigmoid -> d_out [128 x 30000]
  k_gemm_nt_sig<<<dim3((1875 + 3) / 4, 128 / 16), 128, 0, stream>>>(
      xf16, 416, lf, 416, out, 30000, 128, 30000, 416);
}
